// MultiHeadSelfAttention_62904091017969
// MI455X (gfx1250) — matrix operand order repaired and checked by reading, not compiled, not run on hardware
//
#include <hip/hip_runtime.h>
#include <hip/hip_bf16.h>

#define D_MODEL   1024
#define NUM_HEADS 16
#define HEAD_DIM  64
#define SEQ       2048
#define BATCH     4
#define MROWS     (BATCH * SEQ)   // 8192

typedef _Float16 v16h __attribute__((ext_vector_type(16)));
typedef _Float16 v8h  __attribute__((ext_vector_type(8)));
typedef _Float16 v4h  __attribute__((ext_vector_type(4)));
typedef float    v8f  __attribute__((ext_vector_type(8)));

union HF16 { v16h v; v8h h[2]; };

// ---------------------------------------------------------------------------
// DPP-based 16-lane row reductions (no LDS): butterfly over each DPP row of 16
// lanes, which matches the WMMA C/D layout's lane halves exactly.
//   0x0B1 = quad_perm(1,0,3,2)  -> lane ^ 1
//   0x04E = quad_perm(2,3,0,1)  -> lane ^ 2
//   0x141 = row_half_mirror     -> pairs the two quads within 8 lanes
//   0x140 = row_mirror          -> pairs the two 8-groups within 16 lanes
// ---------------------------------------------------------------------------
template <int CTRL>
__device__ __forceinline__ float dpp_f(float x) {
  return __int_as_float(
      __builtin_amdgcn_update_dpp(0, __float_as_int(x), CTRL, 0xf, 0xf, true));
}
__device__ __forceinline__ float rowmax16(float v) {
  v = fmaxf(v, dpp_f<0x0B1>(v));
  v = fmaxf(v, dpp_f<0x04E>(v));
  v = fmaxf(v, dpp_f<0x141>(v));
  v = fmaxf(v, dpp_f<0x140>(v));
  return v;
}
__device__ __forceinline__ float rowsum16(float v) {
  v += dpp_f<0x0B1>(v);
  v += dpp_f<0x04E>(v);
  v += dpp_f<0x141>(v);
  v += dpp_f<0x140>(v);
  return v;
}

// ---------------------------------------------------------------------------
// WMMA fragment loaders (layouts per CDNA5 ISA §7.12.2, wave32)
// ---------------------------------------------------------------------------

// 16-bit A matrix 16x32 (MxK), A row-major with leading dim lda.
__device__ __forceinline__ v16h load_a_frag(const _Float16* A, int lda, int k0) {
  int lane = threadIdx.x & 31;
  int m = lane & 15;
  int g = lane >> 4;
  const _Float16* p = A + (size_t)m * lda + k0 + g * 8;
  HF16 f;
  f.h[0] = *(const v8h*)(p);
  f.h[1] = *(const v8h*)(p + 16);
  return f.v;
}

// 16-bit B matrix 32x16 (KxN) supplied as Bt row-major [N][K] (i.e. B^T).
__device__ __forceinline__ v16h load_bt_frag(const _Float16* Bt, int ldb, int k0) {
  int lane = threadIdx.x & 31;
  int n = lane & 15;
  int g = lane >> 4;
  const _Float16* p = Bt + (size_t)n * ldb + k0 + g * 8;
  HF16 f;
  f.h[0] = *(const v8h*)(p);
  f.h[1] = *(const v8h*)(p + 16);
  return f.v;
}

__device__ __forceinline__ v8f wmma_f16(v16h a, v16h b, v8f c) {
  return __builtin_amdgcn_wmma_f32_16x16x32_f16(false, a, false, b, (short)0, c,
                                                false, false);
}

// ---------------------------------------------------------------------------
// f32 -> f16 conversion (vectorized x4)
// ---------------------------------------------------------------------------
__global__ __launch_bounds__(256)
void cvt_f32_f16_x4(const float4* __restrict__ in, v4h* __restrict__ out, int n4) {
  int i = blockIdx.x * blockDim.x + threadIdx.x;
  int stride = gridDim.x * blockDim.x;
  for (; i < n4; i += stride) {
    float4 a = in[i];
    v4h o;
    o[0] = (_Float16)a.x; o[1] = (_Float16)a.y;
    o[2] = (_Float16)a.z; o[3] = (_Float16)a.w;
    out[i] = o;
  }
}

// ---------------------------------------------------------------------------
// GEMM: Y[M][N] = A[M][K](f16, row-major) * W[N][K]^T (f16, row-major)
// 8 waves/block; each wave owns a 64x64 tile (4x4 WMMA tiles: 16 WMMAs per
// 8 fragment loads -> 32 FLOP/B from L2). Block tile = 256x128.
// MODE: 0 = f32 row-major, 1 = f16 row-major, 2 = f16 [b][h][d][s] (V^T).
// ---------------------------------------------------------------------------
template <int MODE>
__device__ __forceinline__ void store_tile(void* out, v8f c, int m0, int n0, int N) {
  int lane = threadIdx.x & 31;
  int n = n0 + (lane & 15);
  int mbase = m0 + (lane >> 4) * 8;
  if (MODE == 2) {
    // lane holds 8 consecutive s-positions of one (b,h,d) column
    int b = mbase >> 11, s = mbase & (SEQ - 1);
    int hh = n >> 6, d = n & (HEAD_DIM - 1);
    v8h o;
#pragma unroll
    for (int r = 0; r < 8; ++r) o[r] = (_Float16)c[r];
    *(v8h*)&((_Float16*)out)[((size_t)((b * NUM_HEADS + hh) * HEAD_DIM + d)) * SEQ + s] = o;
    return;
  }
#pragma unroll
  for (int r = 0; r < 8; ++r) {
    int m = mbase + r;
    if (MODE == 0) {
      ((float*)out)[(size_t)m * N + n] = c[r];
    } else {
      ((_Float16*)out)[(size_t)m * N + n] = (_Float16)c[r];
    }
  }
}

template <int MODE>
__global__ __launch_bounds__(256)
void gemm_f16_wmma(const _Float16* __restrict__ A, const _Float16* __restrict__ W,
                   void* __restrict__ out, int M, int N, int K) {
  int wave = threadIdx.x >> 5;
  int wm = wave & 3;   // 4 waves along M
  int wn = wave >> 2;  // 2 waves along N
  int m0 = blockIdx.x * 256 + wm * 64;
  int n0 = blockIdx.y * 128 + wn * 64;

  const _Float16* Arow[4];
  const _Float16* Wrow[4];
#pragma unroll
  for (int i = 0; i < 4; ++i) {
    Arow[i] = A + (size_t)(m0 + 16 * i) * K;
    Wrow[i] = W + (size_t)(n0 + 16 * i) * K;
  }

  v8f acc[4][4] = {};

  for (int k0 = 0; k0 < K; k0 += 32) {
    v16h af[4], bf[4];
#pragma unroll
    for (int i = 0; i < 4; ++i) af[i] = load_a_frag(Arow[i], K, k0);
#pragma unroll
    for (int j = 0; j < 4; ++j) bf[j] = load_bt_frag(Wrow[j], K, k0);
#pragma unroll
    for (int i = 0; i < 4; ++i)
#pragma unroll
      for (int j = 0; j < 4; ++j)
        acc[i][j] = wmma_f16(af[i], bf[j], acc[i][j]);
  }

#pragma unroll
  for (int i = 0; i < 4; ++i)
#pragma unroll
    for (int j = 0; j < 4; ++j)
      store_tile<MODE>(out, acc[i][j], m0 + 16 * i, n0 + 16 * j, N);
}

// ---------------------------------------------------------------------------
// Flash attention: one wave per (b, h, 16-row q tile). Streams K/V in chunks
// of 64 rows with base-2 online softmax (DPP row-max; per-lane partial
// denominator reduced once at the end). Q/K are f16 [b][s][h*64+d]; V is f16
// transposed [b][h][d][s]; output Oh is f16 [b][s][h*64+d].
// ---------------------------------------------------------------------------
__global__ __launch_bounds__(128)
void flash_attn_wmma(const _Float16* __restrict__ Qh, const _Float16* __restrict__ Kh,
                     const _Float16* __restrict__ VT, _Float16* __restrict__ Oh) {
  __shared__ __align__(16) _Float16 plds[4][16][64];  // per-wave P staging

  int wave = threadIdx.x >> 5;
  int lane = threadIdx.x & 31;
  int wid = blockIdx.x * 4 + wave;
  int qt = wid & (SEQ / 16 - 1);          // 0..127
  int h  = (wid >> 7) & (NUM_HEADS - 1);  // 0..15
  int b  = wid >> 11;                     // 0..3

  const _Float16* Qbase =
      Qh + ((size_t)(b * SEQ + qt * 16)) * D_MODEL + h * HEAD_DIM;
  const _Float16* Kbase = Kh + ((size_t)b * SEQ) * D_MODEL + h * HEAD_DIM;
  const _Float16* Vbase = VT + ((size_t)(b * NUM_HEADS + h) * HEAD_DIM) * SEQ;

  // Q fragments (held in registers for the whole stream): K-dim 0..31, 32..63
  v16h qf0 = load_a_frag(Qbase, D_MODEL, 0);
  v16h qf1 = load_a_frag(Qbase, D_MODEL, 32);

  v8f o0 = {}, o1 = {}, o2 = {}, o3 = {};
  float mi[8], li[8];  // base-2 running max / per-lane partial denominator
#pragma unroll
  for (int r = 0; r < 8; ++r) { mi[r] = -1e30f; li[r] = 0.0f; }

  int g = lane >> 4;
  int ncol = lane & 15;
  int q_last = qt * 16 + 15;
  // 1/sqrt(HEAD_DIM) * log2(e): softmax computed in base 2 (HW exp is base-2)
  const float sm_scale2 = 0.125f * 1.44269504088896340736f;

  auto chunk = [&](int k0, bool do_mask) {
    // ---- scores: four 16x16 tiles (cols k0+16t .. k0+16t+15) ----
    v8f st[4];
#pragma unroll
    for (int t = 0; t < 4; ++t) {
      const _Float16* kb = Kbase + (size_t)(k0 + 16 * t) * D_MODEL;
      v8f s = {};
      s = wmma_f16(qf0, load_bt_frag(kb, D_MODEL, 0), s);
      s = wmma_f16(qf1, load_bt_frag(kb, D_MODEL, 32), s);
      st[t] = s;
    }
    // ---- scale to log2 domain (+ causal mask on diagonal chunk) ----
    float p[4][8];
#pragma unroll
    for (int t = 0; t < 4; ++t)
#pragma unroll
      for (int r = 0; r < 8; ++r) {
        float v = st[t][r] * sm_scale2;
        if (do_mask) {
          int qrow = qt * 16 + g * 8 + r;
          if (k0 + 16 * t + ncol > qrow) v = -1e30f;
        }
        p[t][r] = v;
      }
    // ---- online softmax: DPP row-max, per-lane partial denominator ----
#pragma unroll
    for (int r = 0; r < 8; ++r) {
      float v = fmaxf(fmaxf(p[0][r], p[1][r]), fmaxf(p[2][r], p[3][r]));
      float mnew = fmaxf(mi[r], rowmax16(v));
      float alpha = exp2f(mi[r] - mnew);
      mi[r] = mnew;
      float e0 = exp2f(p[0][r] - mnew);
      float e1 = exp2f(p[1][r] - mnew);
      float e2 = exp2f(p[2][r] - mnew);
      float e3 = exp2f(p[3][r] - mnew);
      p[0][r] = e0; p[1][r] = e1; p[2][r] = e2; p[3][r] = e3;
      li[r] = li[r] * alpha + ((e0 + e1) + (e2 + e3));
      o0[r] *= alpha; o1[r] *= alpha;
      o2[r] *= alpha; o3[r] *= alpha;
    }
    // ---- stage P through LDS: D-layout -> A-layout (16x64 f16) ----
#pragma unroll
    for (int t = 0; t < 4; ++t)
#pragma unroll
      for (int r = 0; r < 8; ++r)
        plds[wave][g * 8 + r][16 * t + ncol] = (_Float16)p[t][r];
    v16h pf0, pf1;
    {
      const _Float16* row = &plds[wave][lane & 15][0];
      HF16 f;
      f.h[0] = *(const v8h*)(row + g * 8);
      f.h[1] = *(const v8h*)(row + 16 + g * 8);
      pf0 = f.v;
      f.h[0] = *(const v8h*)(row + 32 + g * 8);
      f.h[1] = *(const v8h*)(row + 48 + g * 8);
      pf1 = f.v;
    }
    // ---- O += P @ V over 64 k-rows (V^T gives contiguous B fragments) ----
    o0 = wmma_f16(pf0, load_bt_frag(Vbase + (size_t)(0 * 16) * SEQ, SEQ, k0),      o0);
    o0 = wmma_f16(pf1, load_bt_frag(Vbase + (size_t)(0 * 16) * SEQ, SEQ, k0 + 32), o0);
    o1 = wmma_f16(pf0, load_bt_frag(Vbase + (size_t)(1 * 16) * SEQ, SEQ, k0),      o1);
    o1 = wmma_f16(pf1, load_bt_frag(Vbase + (size_t)(1 * 16) * SEQ, SEQ, k0 + 32), o1);
    o2 = wmma_f16(pf0, load_bt_frag(Vbase + (size_t)(2 * 16) * SEQ, SEQ, k0),      o2);
    o2 = wmma_f16(pf1, load_bt_frag(Vbase + (size_t)(2 * 16) * SEQ, SEQ, k0 + 32), o2);
    o3 = wmma_f16(pf0, load_bt_frag(Vbase + (size_t)(3 * 16) * SEQ, SEQ, k0),      o3);
    o3 = wmma_f16(pf1, load_bt_frag(Vbase + (size_t)(3 * 16) * SEQ, SEQ, k0 + 32), o3);
  };

  // chunks strictly below the diagonal need no mask; only the last does
  int kdiag = q_last & ~63;
  for (int k0 = 0; k0 < kdiag; k0 += 64) chunk(k0, false);
  chunk(kdiag, true);

  // ---- epilogue: single denominator reduction, normalize, store f16 ----
  size_t orow0 = ((size_t)(b * SEQ + qt * 16)) * D_MODEL + h * HEAD_DIM;
#pragma unroll
  for (int r = 0; r < 8; ++r) {
    float inv = 1.0f / rowsum16(li[r]);
    size_t base = orow0 + (size_t)(g * 8 + r) * D_MODEL;
    Oh[base +  0 + ncol] = (_Float16)(o0[r] * inv);
    Oh[base + 16 + ncol] = (_Float16)(o1[r] * inv);
    Oh[base + 32 + ncol] = (_Float16)(o2[r] * inv);
    Oh[base + 48 + ncol] = (_Float16)(o3[r] * inv);
  }
}

// ---------------------------------------------------------------------------
// Host-side orchestration. Workspace layout (bytes, total 88 MiB):
//   [0,16M)   xh    f16 x            [8192][1024]
//   [16M,18M) wqh   f16 Wq           [1024][1024]
//   [18M,20M) wkh   f16 Wk
//   [20M,22M) wvh   f16 Wv
//   [22M,24M) woh   f16 Wo
//   [24M,40M) qh    f16 Q            [b][s][h*64+d]
//   [40M,56M) kh    f16 K            [b][s][h*64+d]
//   [56M,72M) vT    f16 V^T          [b][h][d][s]
//   [72M,88M) oh    f16 attn output  [b][s][h*64+d]
// ---------------------------------------------------------------------------
extern "C" void kernel_launch(void* const* d_in, const int* in_sizes, int n_in,
                              void* d_out, int out_size, void* d_ws, size_t ws_size,
                              hipStream_t stream) {
  const float* x  = (const float*)d_in[0];
  const float* wq = (const float*)d_in[1];
  const float* wk = (const float*)d_in[2];
  const float* wv = (const float*)d_in[3];
  const float* wo = (const float*)d_in[4];
  float* out = (float*)d_out;

  char* ws = (char*)d_ws;
  const size_t MiB = 1u << 20;
  _Float16* xh  = (_Float16*)(ws + 0 * MiB);
  _Float16* wqh = (_Float16*)(ws + 16 * MiB);
  _Float16* wkh = (_Float16*)(ws + 18 * MiB);
  _Float16* wvh = (_Float16*)(ws + 20 * MiB);
  _Float16* woh = (_Float16*)(ws + 22 * MiB);
  _Float16* qh  = (_Float16*)(ws + 24 * MiB);
  _Float16* kh  = (_Float16*)(ws + 40 * MiB);
  _Float16* vT  = (_Float16*)(ws + 56 * MiB);
  _Float16* oh  = (_Float16*)(ws + 72 * MiB);

  // 1) f32 -> f16 converts
  cvt_f32_f16_x4<<<2048, 256, 0, stream>>>((const float4*)x, (v4h*)xh,
                                           MROWS * D_MODEL / 4);
  cvt_f32_f16_x4<<<512, 256, 0, stream>>>((const float4*)wq, (v4h*)wqh,
                                          D_MODEL * D_MODEL / 4);
  cvt_f32_f16_x4<<<512, 256, 0, stream>>>((const float4*)wk, (v4h*)wkh,
                                          D_MODEL * D_MODEL / 4);
  cvt_f32_f16_x4<<<512, 256, 0, stream>>>((const float4*)wv, (v4h*)wvh,
                                          D_MODEL * D_MODEL / 4);
  cvt_f32_f16_x4<<<512, 256, 0, stream>>>((const float4*)wo, (v4h*)woh,
                                          D_MODEL * D_MODEL / 4);

  // 2) projections (block tile 256x128, wave tile 64x64)
  dim3 gg(MROWS / 256, D_MODEL / 128);
  gemm_f16_wmma<1><<<gg, 256, 0, stream>>>(xh, wqh, qh, MROWS, D_MODEL, D_MODEL);
  gemm_f16_wmma<1><<<gg, 256, 0, stream>>>(xh, wkh, kh, MROWS, D_MODEL, D_MODEL);
  gemm_f16_wmma<2><<<gg, 256, 0, stream>>>(xh, wvh, vT, MROWS, D_MODEL, D_MODEL);

  // 3) fused causal attention (one wave per (b, h, 16-row q tile))
  int nwaves = BATCH * NUM_HEADS * (SEQ / 16);  // 8192
  flash_attn_wmma<<<nwaves / 4, 128, 0, stream>>>(qh, kh, vT, oh);

  // 4) output projection (f32 result)
  gemm_f16_wmma<0><<<gg, 256, 0, stream>>>(oh, woh, (void*)out, MROWS, D_MODEL,
                                           D_MODEL);
}
